// VQVAELayer_34205119545780
// MI455X (gfx1250) — matrix operand order repaired and checked by reading, not compiled, not run on hardware
//
#include <hip/hip_runtime.h>

// ---------------------------------------------------------------------------
// VQ-VAE quantization for MI455X (gfx1250, wave32, WMMA + async-to-LDS)
//   inputs:  d_in[0] = inputs     [64,512,256] f32   (N=32768 rows, D=256)
//            d_in[1] = emb_weight [8192,256]   f32   (K codes)
//   outputs: d_out = [ quantized (N*256 f32) | loss (1 f32) | indices (N f32) ]
// ---------------------------------------------------------------------------

typedef __bf16 bf16_t;
typedef __attribute__((ext_vector_type(16))) __bf16 v16bf;
typedef __attribute__((ext_vector_type(8)))  __bf16 v8bf;
typedef __attribute__((ext_vector_type(8)))  float  v8f;

#define KCODES      8192
#define DDIM        256
#define NROWS       32768           // 64*512
#define TILE_CODES  32              // codes staged per LDS tile
#define NTILES      (KCODES / TILE_CODES)   // 256
#define LDS_STRIDE  528             // 256*2 bytes + 16B pad (bank spread)
#define WAVES_WG    8
#define ROWS_WG     (WAVES_WG * 16) // 128 rows per workgroup
#define QELEMS      (NROWS * DDIM)  // 8388608

// ---------------------------------------------------------------------------
// Kernel 1: codebook -> bf16 (RNE) + per-code squared L2 norm
// ---------------------------------------------------------------------------
__global__ void vq_prep_codes(const float* __restrict__ emb,
                              bf16_t* __restrict__ codes_bf,
                              float* __restrict__ code_sq)
{
    const int row = blockIdx.x;      // 0..8191
    const int t   = threadIdx.x;     // 0..255
    const float v = emb[(size_t)row * DDIM + t];
    codes_bf[(size_t)row * DDIM + t] = (bf16_t)v;

    __shared__ float red[256];
    red[t] = v * v;
    __syncthreads();
    for (int s = 128; s > 0; s >>= 1) {
        if (t < s) red[t] += red[t + s];
        __syncthreads();
    }
    if (t == 0) code_sq[row] = red[0];
}

// ---------------------------------------------------------------------------
// Async staging: 4 x b128 per thread, global -> LDS, tracked by ASYNCcnt.
// IOFFSET applies to BOTH the global and LDS address (ISA 10.? async pseudo-
// code), so one base pair + offset:0/16/32/48 covers the thread's 64 bytes.
// ---------------------------------------------------------------------------
__device__ __forceinline__ void async_stage64B(const unsigned char* gsrc,
                                               unsigned lds_addr)
{
    asm volatile(
        "global_load_async_to_lds_b128 %0, %1, off\n\t"
        "global_load_async_to_lds_b128 %0, %1, off offset:16\n\t"
        "global_load_async_to_lds_b128 %0, %1, off offset:32\n\t"
        "global_load_async_to_lds_b128 %0, %1, off offset:48"
        :: "v"(lds_addr), "v"(gsrc) : "memory");
}

__device__ __forceinline__ void wait_async0()
{
    asm volatile("s_wait_asynccnt 0x0" ::: "memory");
}

// ---------------------------------------------------------------------------
// Kernel 2: WMMA distance + streaming argmin
//   grid = 256 WGs x 256 thr (8 waves); wave w owns rows [wg*128+w*16, +16)
//   A fragments (activations) held in VGPRs for the entire K loop.
//   Codebook tiles double-buffered through LDS via async-to-LDS DMA.
// ---------------------------------------------------------------------------
__global__ __launch_bounds__(256, 1)
void vq_argmin(const float*  __restrict__ x,
               const bf16_t* __restrict__ codes,
               const float*  __restrict__ code_sq,
               int* __restrict__ out_idx)
{
    __shared__ unsigned char lds[2][TILE_CODES * LDS_STRIDE];

    const int tid  = threadIdx.x;
    const int lane = tid & 31;
    const int wave = tid >> 5;
    const int h    = lane >> 4;     // 16-lane half
    const int l16  = lane & 15;

    const int row_base = blockIdx.x * ROWS_WG + wave * 16;
    const int row      = row_base + l16;

    // staging geometry: 8 threads per code row, 64B per thread
    const int prow = tid >> 3;          // 0..31 code row within tile
    const int pcol = (tid & 7) * 64;    // byte offset within 512B row
    const unsigned char* gbase = (const unsigned char*)codes + prow * 512 + pcol;
    const unsigned ldsa0 = (unsigned)(uintptr_t)&lds[0][prow * LDS_STRIDE + pcol];
    const unsigned ldsa1 = (unsigned)(uintptr_t)&lds[1][prow * LDS_STRIDE + pcol];

    // kick off tile 0 DMA; it runs behind the A-fragment load/convert below
    async_stage64B(gbase, ldsa0);

    // ---- A fragments: ISA 16-bit A 16x32 layout ----------------------------
    // lanes 0-15 : elems 0-7 = K[kt*32+0..7],  elems 8-15 = K[kt*32+16..23]
    // lanes16-31 : elems 0-7 = K[kt*32+8..15], elems 8-15 = K[kt*32+24..31]
    v16bf a[8];
    {
        const float* xr = x + (size_t)row * DDIM;
        #pragma unroll
        for (int kt = 0; kt < 8; ++kt) {
            const float4 f0 = *(const float4*)(xr + kt * 32 + h * 8 + 0);
            const float4 f1 = *(const float4*)(xr + kt * 32 + h * 8 + 4);
            const float4 f2 = *(const float4*)(xr + kt * 32 + 16 + h * 8 + 0);
            const float4 f3 = *(const float4*)(xr + kt * 32 + 16 + h * 8 + 4);
            v16bf av;
            av[0]  = (bf16_t)f0.x; av[1]  = (bf16_t)f0.y;
            av[2]  = (bf16_t)f0.z; av[3]  = (bf16_t)f0.w;
            av[4]  = (bf16_t)f1.x; av[5]  = (bf16_t)f1.y;
            av[6]  = (bf16_t)f1.z; av[7]  = (bf16_t)f1.w;
            av[8]  = (bf16_t)f2.x; av[9]  = (bf16_t)f2.y;
            av[10] = (bf16_t)f2.z; av[11] = (bf16_t)f2.w;
            av[12] = (bf16_t)f3.x; av[13] = (bf16_t)f3.y;
            av[14] = (bf16_t)f3.z; av[15] = (bf16_t)f3.w;
            a[kt] = av;
        }
    }

    float best[8]; int bidx[8];
    #pragma unroll
    for (int r = 0; r < 8; ++r) { best[r] = 3.4e38f; bidx[r] = 0; }

    for (int tile = 0; tile < NTILES; ++tile) {
        // tile's data (issued last iteration / pre-loop) has landed in LDS:
        wait_async0();
        __syncthreads();   // all waves: compute(t-1) done, tile t visible

        // issue DMA for tile+1 into the buffer freed by compute(t-1)
        if (tile + 1 < NTILES) {
            async_stage64B(gbase + (size_t)(tile + 1) * (TILE_CODES * 512),
                           ((tile + 1) & 1) ? ldsa1 : ldsa0);
        }

        const unsigned char* buf = lds[tile & 1];

        // ---- two 16-code subtiles: 8 WMMAs each ---------------------------
        #pragma unroll
        for (int s = 0; s < 2; ++s) {
            v8f acc = {};
            // B layout (32x16 bf16): lane = column(code), lanes0-15 K[kt*32+0..7,16..23],
            // lanes16-31 K[kt*32+8..15,24..31] -> matches A fragment k order.
            const unsigned char* bp =
                buf + (s * 16 + l16) * LDS_STRIDE + h * 16;
            #pragma unroll
            for (int kt = 0; kt < 8; ++kt) {
                const v8bf blo = *(const v8bf*)(bp + kt * 64);
                const v8bf bhi = *(const v8bf*)(bp + kt * 64 + 32);
                const v16bf bfrag = __builtin_shufflevector(
                    blo, bhi, 0,1,2,3,4,5,6,7,8,9,10,11,12,13,14,15);
                acc = __builtin_amdgcn_wmma_f32_16x16x32_bf16(
                    false, a[kt], false, bfrag, (short)0, acc, false, false);
            }
            const int code = tile * TILE_CODES + s * 16 + l16;
            const float sq = code_sq[code];
            #pragma unroll
            for (int r = 0; r < 8; ++r) {
                // dist = -2*x.e + |e|^2   (|x|^2 is row-constant, drops out)
                const float d = fmaf(-2.0f, acc[r], sq);
                if (d < best[r]) { best[r] = d; bidx[r] = code; }
            }
        }
    }

    // ---- cross-lane argmin; C layout: VGPR r, half h -> row r+8h ----------
    #pragma unroll
    for (int r = 0; r < 8; ++r) {
        float d = best[r]; int i = bidx[r];
        #pragma unroll
        for (int off = 8; off >= 1; off >>= 1) {
            const float od = __shfl_xor(d, off, 32);
            const int   oi = __shfl_xor(i, off, 32);
            if (od < d || (od == d && oi < i)) { d = od; i = oi; }
        }
        if (l16 == 0) out_idx[row_base + 8 * h + r] = i;
    }
}

// ---------------------------------------------------------------------------
// Kernel 3: gather nearest code (f32 codebook), per-row squared error,
//           emit indices as float (output dtype)
// ---------------------------------------------------------------------------
__global__ void vq_gather(const float* __restrict__ x,
                          const float* __restrict__ emb,
                          const int*   __restrict__ idx,
                          float* __restrict__ out_q,
                          float* __restrict__ out_ind,
                          float* __restrict__ partial)
{
    const int row = blockIdx.x;
    const int t   = threadIdx.x;
    const int code = idx[row];
    const float q  = emb[(size_t)code * DDIM + t];
    const float xv = x[(size_t)row * DDIM + t];
    out_q[(size_t)row * DDIM + t] = q;   // straight-through fwd value == q

    const float diff = q - xv;
    __shared__ float red[256];
    red[t] = diff * diff;
    __syncthreads();
    for (int s = 128; s > 0; s >>= 1) {
        if (t < s) red[t] += red[t + s];
        __syncthreads();
    }
    if (t == 0) {
        partial[row]  = red[0];
        out_ind[row]  = (float)code;
    }
}

// ---------------------------------------------------------------------------
// Kernel 4: deterministic final loss reduction: loss = 2 * mean((q-x)^2)
// ---------------------------------------------------------------------------
__global__ void vq_loss(const float* __restrict__ partial,
                        float* __restrict__ out_loss)
{
    __shared__ float red[256];
    const int t = threadIdx.x;
    float s = 0.0f;
    for (int i = t; i < NROWS; i += 256) s += partial[i];
    red[t] = s;
    __syncthreads();
    for (int k = 128; k > 0; k >>= 1) {
        if (t < k) red[t] += red[t + k];
        __syncthreads();
    }
    if (t == 0) out_loss[0] = 2.0f * red[0] / (float)QELEMS;
}

// ---------------------------------------------------------------------------
extern "C" void kernel_launch(void* const* d_in, const int* in_sizes, int n_in,
                              void* d_out, int out_size, void* d_ws, size_t ws_size,
                              hipStream_t stream)
{
    const float* x   = (const float*)d_in[0];   // [64,512,256]
    const float* emb = (const float*)d_in[1];   // [8192,256]

    float* out      = (float*)d_out;
    float* out_q    = out;                      // QELEMS floats
    float* out_loss = out + QELEMS;             // 1 float
    float* out_ind  = out + QELEMS + 1;         // NROWS floats

    unsigned char* ws = (unsigned char*)d_ws;
    bf16_t* codes_bf = (bf16_t*)ws;                                   // 4 MB
    float*  code_sq  = (float*)(ws + (size_t)4 * 1024 * 1024);        // 32 KB
    int*    idx      = (int*)  (ws + (size_t)4 * 1024 * 1024 + 32768);// 128 KB
    float*  partial  = (float*)(ws + (size_t)4 * 1024 * 1024 + 32768
                                   + (size_t)NROWS * 4);              // 128 KB

    vq_prep_codes<<<KCODES, 256, 0, stream>>>(emb, codes_bf, code_sq);
    vq_argmin<<<NROWS / ROWS_WG, 256, 0, stream>>>(x, codes_bf, code_sq, idx);
    vq_gather<<<NROWS, 256, 0, stream>>>(x, emb, idx, out_q, out_ind, partial);
    vq_loss<<<1, 256, 0, stream>>>(partial, out_loss);
}
